// CortexModel_49383533969460
// MI455X (gfx1250) — matrix operand order repaired and checked by reading, not compiled, not run on hardware
//
#include <hip/hip_runtime.h>
#include <hip/hip_bf16.h>

// ---------------------------------------------------------------------------
// CortexModel on MI455X (gfx1250, wave32)
//   - afferent: memory-bound 255MB weight stream -> f32 VALU + wave32 reduce
//   - lateral convs: Toeplitz-band WMMA (v_wmma_f32_16x16x32_bf16)
// ---------------------------------------------------------------------------

typedef __attribute__((ext_vector_type(16))) __bf16 v16bf;
typedef __attribute__((ext_vector_type(8)))  float  v8f;

#define Cc     3
#define Xn     192
#define Yn     192
#define RF     24
#define IMG    224
#define EXK    38
#define INK    94
#define ROWP   320              // padded-row pitch (covers r_base + 127 < 320)
#define PCOLS  286              // padded image size (192 + 2*47)
#define NPIX   (Xn * Yn)        // 36864

// round-to-nearest-even f32 -> bf16 bits
__device__ __forceinline__ unsigned short f2bf(float f) {
    unsigned int u = __float_as_uint(f);
    u = (u + 0x7FFFu + ((u >> 16) & 1u)) >> 16;
    return (unsigned short)u;
}

// ---------------------------------------------------------------------------
// 1) padded, transposed (column-major) bf16 activation image:
//    padT[c][col][row], row pitch ROWP, zero outside valid region.
// ---------------------------------------------------------------------------
__global__ void k_prep_pad(const float* __restrict__ old_act,
                           unsigned short* __restrict__ padT) {
    int idx = blockIdx.x * blockDim.x + threadIdx.x;
    const int TOT = Cc * PCOLS * ROWP;
    if (idx >= TOT) return;
    int row = idx % ROWP;
    int t   = idx / ROWP;
    int col = t % PCOLS;
    int c   = t / PCOLS;
    float v = 0.f;
    if (row >= 47 && row < 239 && col >= 47 && col < 239)
        v = old_act[((size_t)c * Xn + (row - 47)) * Yn + (col - 47)];
    padT[idx] = f2bf(v);
}

// ---------------------------------------------------------------------------
// 2) Toeplitz A-fragment tables, pre-swizzled into WMMA bf16 A fragments (contiguous k, as B uses):
//    A[r,k] = (W*mask)[k - r, v]  for k-r in [0,U), else 0.
//    Table layout: [frag = v*NCH + chunk][lane][16 bf16]  (32B per lane)
//    Per-lane element j -> K = j + (lane<16 ? 0 : 16), M = lane%16
// ---------------------------------------------------------------------------
__global__ void k_prep_A(const float* __restrict__ exw, const float* __restrict__ inw,
                         const float* __restrict__ exm, const float* __restrict__ inm,
                         unsigned short* __restrict__ Aex,
                         unsigned short* __restrict__ Ain) {
    int tid  = blockIdx.x * blockDim.x + threadIdx.x;
    int lane = tid & 31;
    int fid  = tid >> 5;
    const int EXF = EXK * 2;           // 38 v-shifts * 2 K-chunks
    const int INF = INK * 4;           // 94 v-shifts * 4 K-chunks
    const float* w; const float* m; unsigned short* out; int U, NCH;
    if (fid < EXF)            { U = EXK; NCH = 2; w = exw; m = exm; out = Aex + fid * 512 + lane * 16; }
    else if (fid < EXF + INF) { fid -= EXF;
                                U = INK; NCH = 4; w = inw; m = inm; out = Ain + fid * 512 + lane * 16; }
    else return;
    int v     = fid / NCH;
    int chunk = fid % NCH;
    int r     = lane & 15;
    int khi   = (lane & 16) ? 16 : 0;
#pragma unroll
    for (int j = 0; j < 16; ++j) {
        int K  = j + khi;
        int u  = chunk * 32 + K - r;
        float val = (u >= 0 && u < U) ? w[u * U + v] * m[u * U + v] : 0.f;
        out[j] = f2bf(val);
    }
}

// ---------------------------------------------------------------------------
// 3) afferent: one wave32 per output pixel; streams 3*576 f32 weights
//    (coalesced, lane-strided), multiplies by mask and gathered input patch,
//    wave shuffle reduction. Pure HBM-bandwidth kernel (~255MB).
// ---------------------------------------------------------------------------
__global__ void __launch_bounds__(256)
k_afferent(const float* __restrict__ input,
           const float* __restrict__ aw,
           const float* __restrict__ amask,
           const int*   __restrict__ rfs,
           float*       __restrict__ aff) {
    int wave = (blockIdx.x * blockDim.x + threadIdx.x) >> 5;
    int lane = threadIdx.x & 31;
    if (wave >= NPIX) return;
    int sx = rfs[wave * 2 + 0];
    int sy = rfs[wave * 2 + 1];
    float acc = 0.f;
#pragma unroll
    for (int c = 0; c < Cc; ++c) {
        const float* wp = aw + ((size_t)c * NPIX + wave) * (RF * RF);
        const float* ip = input + (size_t)c * IMG * IMG;
        for (int t = lane; t < RF * RF; t += 32) {
            int u = t / RF, vv = t - u * RF;
            acc += wp[t] * amask[t] * ip[(sx + u) * IMG + (sy + vv)];
        }
    }
#pragma unroll
    for (int off = 16; off; off >>= 1) acc += __shfl_xor(acc, off, 32);
    if (lane == 0) aff[wave] = acc;
}

// ---------------------------------------------------------------------------
// 4) lateral conv via WMMA. One wave per (channel, 16x16 output tile).
//    For each column shift v and 32-deep K chunk:
//      A = precomputed Toeplitz fragment (32B/lane, coalesced across wave)
//      B = contiguous 32B/lane column slice of the transposed padded image
//      acc = wmma_f32_16x16x32_bf16(A, B, acc)
//    OFF: slice offset into padded coords (28 for exc, 0 for inh).
// ---------------------------------------------------------------------------
union FragU {
    unsigned short s[16];
    unsigned int   u[8];
    uint2          d[4];
    uint4          q[2];
    v16bf          v;
};

template <int U, int NCH, int OFF, bool SUB>
__global__ void __launch_bounds__(128)
k_conv(const unsigned short* __restrict__ padT,
       const unsigned short* __restrict__ Atab,
       const float* __restrict__ aff,
       float* __restrict__ out) {
    int wave = (blockIdx.x * blockDim.x + threadIdx.x) >> 5;
    int lane = threadIdx.x & 31;
    const int NTILE = (Xn / 16) * (Yn / 16);        // 144 per channel
    if (wave >= Cc * NTILE) return;
    int c  = wave / NTILE;
    int t  = wave % NTILE;
    int x0 = (t / (Yn / 16)) * 16;
    int y0 = (t % (Yn / 16)) * 16;

    int nlane = lane & 15;
    int rbase = x0 + OFF + ((lane & 16) ? 16 : 0);   // K-row start for this lane half
    const unsigned short* colbase =
        padT + ((size_t)c * PCOLS + (y0 + OFF + nlane)) * ROWP + rbase;
    const unsigned short* atab = Atab + lane * 16;

    v8f acc = {0.f, 0.f, 0.f, 0.f, 0.f, 0.f, 0.f, 0.f};

    for (int v = 0; v < U; ++v) {
        const unsigned short* bcol = colbase + v * ROWP;   // shift input column by v
#pragma unroll
        for (int ch = 0; ch < NCH; ++ch) {
            FragU a, b;
            const uint4* ap = (const uint4*)(atab + (size_t)(v * NCH + ch) * 512);
            a.q[0] = ap[0];
            a.q[1] = ap[1];
            const unsigned short* bp = bcol + ch * 32;     // contiguous 16 bf16 per lane
            if ((OFF & 7) == 0) {                          // 32B-aligned path (inh)
                const uint4* bq = (const uint4*)bp;
                b.q[0] = bq[0];
                b.q[1] = bq[1];
            } else {                                       // 8B-aligned path (exc)
                const uint2* bd = (const uint2*)bp;
#pragma unroll
                for (int i = 0; i < 4; ++i) b.d[i] = bd[i];
            }
            acc = __builtin_amdgcn_wmma_f32_16x16x32_bf16(
                false, a.v, false, b.v, (short)0, acc, false, false);
        }
    }

    // C/D layout: VGPR i, lane L -> M = i + (L<16?0:8), N = L%16
    int mhi = (lane & 16) ? 8 : 0;
#pragma unroll
    for (int i = 0; i < 8; ++i) {
        int x = x0 + i + mhi;
        int y = y0 + nlane;
        size_t o = ((size_t)c * Xn + x) * Yn + y;
        if (SUB) out[o] -= 0.9f * acc[i];
        else     out[o]  = aff[x * Yn + y] + 0.9f * acc[i];
    }
}

// ---------------------------------------------------------------------------
extern "C" void kernel_launch(void* const* d_in, const int* in_sizes, int n_in,
                              void* d_out, int out_size, void* d_ws, size_t ws_size,
                              hipStream_t stream) {
    const float* input   = (const float*)d_in[0];
    const float* aw      = (const float*)d_in[1];
    const float* exw     = (const float*)d_in[2];
    const float* inw     = (const float*)d_in[3];
    const float* amask   = (const float*)d_in[4];
    const float* exm     = (const float*)d_in[5];
    const float* inm     = (const float*)d_in[6];
    const float* old_act = (const float*)d_in[7];
    const int*   rfs     = (const int*)d_in[8];
    float* out = (float*)d_out;

    // workspace layout (all 32B aligned)
    char* ws = (char*)d_ws;
    float*          aff  = (float*)ws;                               // 147456 B
    unsigned short* padT = (unsigned short*)(ws + 147456);           // 549120 B
    unsigned short* Aex  = (unsigned short*)(ws + 147456 + 549120);  //  77824 B
    unsigned short* Ain  = (unsigned short*)(ws + 147456 + 549120 + 77824); // 385024 B

    {   // pad + transpose + bf16 convert
        int tot = Cc * PCOLS * ROWP;
        k_prep_pad<<<(tot + 255) / 256, 256, 0, stream>>>(old_act, padT);
    }
    {   // Toeplitz A-fragment tables (masked kernels, WMMA A fragments, contiguous k)
        int tot = (EXK * 2 + INK * 4) * 32;
        k_prep_A<<<(tot + 255) / 256, 256, 0, stream>>>(exw, inw, exm, inm, Aex, Ain);
    }
    // afferent: 36864 waves, 8 waves / 256-thread block
    k_afferent<<<NPIX / 8, 256, 0, stream>>>(input, aw, amask, rfs, aff);

    // exc: out = aff + 0.9*exc   (38x38 kernel, 2 K-chunks, slice offset 28)
    k_conv<EXK, 2, 28, false><<<(Cc * 144 + 3) / 4, 128, 0, stream>>>(padT, Aex, aff, out);
    // inh: out -= 0.9*inh        (94x94 kernel, 4 K-chunks, slice offset 0)
    k_conv<INK, 4, 0, true><<<(Cc * 144 + 3) / 4, 128, 0, stream>>>(padT, Ain, aff, out);
}
